// Hierarchical_Relation_Aggregation_71451075937124
// MI455X (gfx1250) — matrix operand order repaired and checked by reading, not compiled, not run on hardware
//
#include <hip/hip_runtime.h>
#include <hip/hip_bf16.h>

// ---------------------------------------------------------------------------
// Problem constants (from the reference setup)
// ---------------------------------------------------------------------------
#define NB      4          // batch
#define MB      48         // detections per batch
#define IMW     96
#define HW      (IMW*IMW)  // 9216 pixels
#define CORP    324        // cor_planes = 4*(2*4+1)^2
#define CORPP   352        // cor_planes padded to K%32==0
#define NM      (NB*MB)    // 192 rois
#define NMP     256        // rois padded to rows%128==0 for the head GEMM
#define NMM     (NM*MB)    // 9216 pairs
#define RFK     (128*49)   // 6272: head GEMM K (rf portion)
#define HLD     (132*49)   // 6468: head weight row stride

typedef __attribute__((ext_vector_type(16))) _Float16 v16h;
typedef __attribute__((ext_vector_type(8)))  float    v8f;
typedef __attribute__((ext_vector_type(4)))  unsigned u32x4;
typedef __attribute__((ext_vector_type(8)))  int      i32x8;
typedef __attribute__((ext_vector_type(4)))  int      i32x4;

#if defined(__has_builtin)
#if __has_builtin(__builtin_amdgcn_tensor_load_to_lds) && \
    __has_builtin(__builtin_amdgcn_s_wait_tensorcnt)
#define HAVE_TDM 1
#endif
#endif

union V16H { v16h v; _Float16 h[16]; };

// ---------------------------------------------------------------------------
// WMMA fragment loaders from LDS tiles (row stride = 32 halfs)
// A (16x32 MxK, f16): lane<16 -> m=lane, k in {0..7, 16..23}
//                     lane>=16 -> m=lane-16, k in {8..15, 24..31}
// B (32x16 KxN, f16) staged transposed (Bt[n][k]): lane<16 -> n=lane, k in {0..7, 16..23}
//                     lane>=16 -> n=lane-16, k in {8..15, 24..31}
// ---------------------------------------------------------------------------
__device__ __forceinline__ v16h load_fragA(const _Float16* __restrict__ row, int lane) {
    const _Float16* p = row + ((lane & 16) ? 8 : 0);
    V16H f;
#pragma unroll
    for (int i = 0; i < 8; ++i) { f.h[i] = p[i]; f.h[8 + i] = p[16 + i]; }
    return f.v;
}

__device__ __forceinline__ v16h load_fragB(const _Float16* __restrict__ row, int lane) {
    const _Float16* p = row + ((lane & 16) ? 8 : 0);
    V16H f;
#pragma unroll
    for (int i = 0; i < 8; ++i) { f.h[i] = p[i]; f.h[8 + i] = p[16 + i]; }
    return f.v;
}

__device__ __forceinline__ v8f wmma_step(v16h a, v16h b, v8f c) {
    return __builtin_amdgcn_wmma_f32_16x16x32_f16(
        false, a, false, b, (short)0, c, false, false);
}

// ---------------------------------------------------------------------------
// Tensor Data Mover: DMA a 2D tile (tile0 x tile1 elements of 2 bytes) from a
// row-major tensor (row length stride0, dim0 x dim1) into LDS at lds_addr.
// Issued by one wave; caller waits tensorcnt and barriers.
// ---------------------------------------------------------------------------
#ifdef HAVE_TDM
__device__ __forceinline__ void tdm_load_2d(unsigned lds_addr, const void* gaddr,
                                            int dim0, int dim1, int stride0,
                                            int tile0, int tile1) {
    unsigned long long ga = (unsigned long long)gaddr;
    u32x4 g0;
    g0[0] = 1u;                                        // count=1
    g0[1] = lds_addr;                                  // lds_addr (bytes)
    g0[2] = (unsigned)ga;                              // global_addr[31:0]
    g0[3] = (unsigned)((ga >> 32) & 0x01ffffffu)       // global_addr[56:32]
          | (2u << 30);                                // type=2 (image)
    i32x8 g1;
    g1[0] = 1 << 16;                                   // data_size=1 (2 bytes)
    g1[1] = (int)(((unsigned)dim0 & 0xffffu) << 16);   // tensor_dim0[15:0]
    g1[2] = (int)(((unsigned)dim0 >> 16) | (((unsigned)dim1 & 0xffffu) << 16));
    g1[3] = (int)(((unsigned)dim1 >> 16) | ((unsigned)tile0 << 16));
    g1[4] = tile1 & 0xffff;                            // tile_dim1 | tile_dim2=0
    g1[5] = stride0;                                   // tensor_dim0_stride[31:0]
    g1[6] = 0;
    g1[7] = 0;
    i32x4 gz = {};
#if __clang_major__ >= 23
    i32x8 gz8 = {};
    __builtin_amdgcn_tensor_load_to_lds(g0, g1, gz, gz, gz8, 0);
#else
    __builtin_amdgcn_tensor_load_to_lds(g0, g1, gz, gz, 0);
#endif
}
#endif

// ---------------------------------------------------------------------------
// f32 -> f16 conversion with inner-dim zero padding:
// dst[r*Kpad + k] = (k < K) ? (f16)src[r*K + k] : 0
// ---------------------------------------------------------------------------
__global__ void cvt_pad_kernel(const float* __restrict__ src,
                               _Float16* __restrict__ dst,
                               int K, int Kpad, long long total) {
    long long idx = (long long)blockIdx.x * blockDim.x + threadIdx.x;
    if (idx >= total) return;
    int r = (int)(idx / Kpad);
    int k = (int)(idx - (long long)r * Kpad);
    dst[idx] = (k < K) ? (_Float16)src[(long long)r * K + k] : (_Float16)0.f;
}

// ---------------------------------------------------------------------------
// Conv-as-GEMM (implicit im2col), NCHW, f16 in/out, f32 bias, relu.
// Block: 256 threads = 8 waves. Block tile: 128 (Cout) x 64 (pixels).
// Wave tile: 32x32 = 2x2 fragments = 4 WMMAs per K-step (A/B frag reuse).
// Weights tile DMA'd by the Tensor Data Mover. K must be a multiple of 32.
// grid: (HW/64, Cout/128, NB)
// ---------------------------------------------------------------------------
template <int KH, int KW, int PAD>
__global__ __launch_bounds__(256)
void conv_gemm_kernel(const _Float16* __restrict__ in,
                      const _Float16* __restrict__ wgt,
                      const float* __restrict__ bias,
                      _Float16* __restrict__ out,
                      int Cin, int Cout, int relu) {
    __shared__ _Float16 As[128 * 32];   // weights tile (m-major)
    __shared__ _Float16 Bs[64 * 32];    // im2col tile, transposed (n-major)

    const int tid  = threadIdx.x;
    const int lane = tid & 31;
    const int wid  = tid >> 5;
    const int m_sub = (wid & 3) * 32;
    const int n_sub = (wid >> 2) * 32;
    const int p0 = blockIdx.x * 64;
    const int m0 = blockIdx.y * 128;
    const int batch = blockIdx.z;
    constexpr int KHW = KH * KW;
    const int K = Cin * KHW;

    in  += (size_t)batch * Cin  * HW;
    out += (size_t)batch * Cout * HW;

    v8f acc00 = {}, acc01 = {}, acc10 = {}, acc11 = {};

    for (int k0 = 0; k0 < K; k0 += 32) {
        // ---- stage A (weights) 128x32 ----
#ifdef HAVE_TDM
        if (wid == 0) {
            tdm_load_2d((unsigned)(size_t)(void*)As,
                        wgt + (size_t)m0 * K + k0, K, Cout, K, 32, 128);
            __builtin_amdgcn_s_wait_tensorcnt(0);
        }
#else
        {
            int idx = tid * 16;
            int m = idx >> 5, kk = idx & 31;
            const _Float16* s = wgt + (size_t)(m0 + m) * K + k0 + kk;
            *(uint4*)&As[idx]     = *(const uint4*)s;
            *(uint4*)&As[idx + 8] = *(const uint4*)(s + 8);
        }
#endif
        // ---- stage B (im2col, transposed Bt[n][k]) 64x32 ----
        if (KHW == 1) {
            // 1x1 conv: 8 contiguous pixels for one k -> one b128 global load
            const int k  = tid >> 3;
            const int n8 = (tid & 7) * 8;
            union { uint4 u; _Float16 h[8]; } pk;
            pk.u = *(const uint4*)&in[(size_t)(k0 + k) * HW + p0 + n8];
#pragma unroll
            for (int i = 0; i < 8; ++i) Bs[(n8 + i) * 32 + k] = pk.h[i];
        } else {
            // 3x3 conv: branch-free clamped gather, one b128 LDS store
            const int idx = tid * 8;
            const int n = idx >> 5;
            const int kk = idx & 31;
            const int pid = p0 + n;
            const int y = pid / IMW, x = pid - y * IMW;
            union { uint4 u; _Float16 h[8]; } pk;
#pragma unroll
            for (int i = 0; i < 8; ++i) {
                int gk = k0 + kk + i;
                int ci = gk / KHW;
                int r  = gk - ci * KHW;
                int ky = r / KW, kx = r - ky * KW;
                int sy = y + ky - PAD, sx = x + kx - PAD;
                bool inb = ((unsigned)sy < (unsigned)IMW) &
                           ((unsigned)sx < (unsigned)IMW);
                int syc = inb ? sy : 0;
                int sxc = inb ? sx : 0;
                _Float16 t = in[(size_t)ci * HW + syc * IMW + sxc];
                pk.h[i] = inb ? t : (_Float16)0.f;
            }
            *(uint4*)&Bs[idx] = pk.u;
        }
        __syncthreads();
        {
            v16h a0 = load_fragA(&As[(m_sub +      (lane & 15)) * 32], lane);
            v16h a1 = load_fragA(&As[(m_sub + 16 + (lane & 15)) * 32], lane);
            v16h b0 = load_fragB(&Bs[(n_sub +      (lane & 15)) * 32], lane);
            v16h b1 = load_fragB(&Bs[(n_sub + 16 + (lane & 15)) * 32], lane);
            acc00 = wmma_step(a0, b0, acc00);
            acc01 = wmma_step(a0, b1, acc01);
            acc10 = wmma_step(a1, b0, acc10);
            acc11 = wmma_step(a1, b1, acc11);
        }
        __syncthreads();
    }

    const int mbase = m0 + m_sub + ((lane & 16) ? 8 : 0);
    const int nbase = p0 + n_sub + (lane & 15);
#pragma unroll
    for (int r = 0; r < 8; ++r) {
        float b_lo = bias[mbase + r], b_hi = bias[mbase + 16 + r];
        float v00 = acc00[r] + b_lo, v01 = acc01[r] + b_lo;
        float v10 = acc10[r] + b_hi, v11 = acc11[r] + b_hi;
        if (relu) {
            v00 = fmaxf(v00, 0.f); v01 = fmaxf(v01, 0.f);
            v10 = fmaxf(v10, 0.f); v11 = fmaxf(v11, 0.f);
        }
        out[(size_t)(mbase + r)      * HW + nbase]      = (_Float16)v00;
        out[(size_t)(mbase + r)      * HW + nbase + 16] = (_Float16)v01;
        out[(size_t)(mbase + 16 + r) * HW + nbase]      = (_Float16)v10;
        out[(size_t)(mbase + 16 + r) * HW + nbase + 16] = (_Float16)v11;
    }
}

// ---------------------------------------------------------------------------
// FC GEMM: Y[rows x O] = X[rows x K] * W[O x K]^T (+bias, optional relu)
// Block tile 128 rows x 64 out-features, 8 waves, 4 WMMAs/wave/K-step.
// X tile DMA'd into LDS by the Tensor Data Mover.
// Requires rows%128==0, O%64==0, K%32==0.
// grid: (rows/128, O/64)
// ---------------------------------------------------------------------------
__global__ __launch_bounds__(256)
void fc_gemm_kernel(const _Float16* __restrict__ X, int lda, int rows,
                    const _Float16* __restrict__ Wt, int ldb,
                    const float* __restrict__ bias,
                    _Float16* __restrict__ Y, int ldo,
                    int K, int relu) {
    __shared__ _Float16 As[128 * 32];
    __shared__ _Float16 Bs[64 * 32];

    const int tid  = threadIdx.x;
    const int lane = tid & 31;
    const int wid  = tid >> 5;
    const int m_sub = (wid & 3) * 32;
    const int n_sub = (wid >> 2) * 32;
    const int r0 = blockIdx.x * 128;
    const int o0 = blockIdx.y * 64;

    v8f acc00 = {}, acc01 = {}, acc10 = {}, acc11 = {};

    for (int k0 = 0; k0 < K; k0 += 32) {
        // ---- stage A (activations) 128x32 ----
#ifdef HAVE_TDM
        if (wid == 0) {
            tdm_load_2d((unsigned)(size_t)(void*)As,
                        X + (size_t)r0 * lda + k0, lda, rows, lda, 32, 128);
            __builtin_amdgcn_s_wait_tensorcnt(0);
        }
#else
        {
            int idx = tid * 16;
            int m = idx >> 5, kk = idx & 31;
            const _Float16* s = X + (size_t)(r0 + m) * lda + k0 + kk;
            *(uint4*)&As[idx]     = *(const uint4*)s;
            *(uint4*)&As[idx + 8] = *(const uint4*)(s + 8);
        }
#endif
        // ---- stage B (weights, transposed) 64x32: two b64 copies ----
        {
            int idx = tid * 8;
            const _Float16* s = Wt + (size_t)(o0 + (idx >> 5)) * ldb + k0 + (idx & 31);
            __builtin_prefetch(s + 32, 0, 1);       // global_prefetch_b8
            *(uint2*)&Bs[idx]     = *(const uint2*)s;
            *(uint2*)&Bs[idx + 4] = *(const uint2*)(s + 4);
        }
        __syncthreads();
        {
            v16h a0 = load_fragA(&As[(m_sub +      (lane & 15)) * 32], lane);
            v16h a1 = load_fragA(&As[(m_sub + 16 + (lane & 15)) * 32], lane);
            v16h b0 = load_fragB(&Bs[(n_sub +      (lane & 15)) * 32], lane);
            v16h b1 = load_fragB(&Bs[(n_sub + 16 + (lane & 15)) * 32], lane);
            acc00 = wmma_step(a0, b0, acc00);
            acc01 = wmma_step(a0, b1, acc01);
            acc10 = wmma_step(a1, b0, acc10);
            acc11 = wmma_step(a1, b1, acc11);
        }
        __syncthreads();
    }

    const int mbase = r0 + m_sub + ((lane & 16) ? 8 : 0);
    const int nbase = o0 + n_sub + (lane & 15);
    float bc0 = bias ? bias[nbase] : 0.f;
    float bc1 = bias ? bias[nbase + 16] : 0.f;
#pragma unroll
    for (int r = 0; r < 8; ++r) {
        float v00 = acc00[r] + bc0, v01 = acc01[r] + bc1;
        float v10 = acc10[r] + bc0, v11 = acc11[r] + bc1;
        if (relu) {
            v00 = fmaxf(v00, 0.f); v01 = fmaxf(v01, 0.f);
            v10 = fmaxf(v10, 0.f); v11 = fmaxf(v11, 0.f);
        }
        Y[(size_t)(mbase + r)      * ldo + nbase]      = (_Float16)v00;
        Y[(size_t)(mbase + r)      * ldo + nbase + 16] = (_Float16)v01;
        Y[(size_t)(mbase + 16 + r) * ldo + nbase]      = (_Float16)v10;
        Y[(size_t)(mbase + 16 + r) * ldo + nbase + 16] = (_Float16)v11;
    }
}

// ---------------------------------------------------------------------------
// ROI align: 192 rois, 128 channels, 7x7 bins, 2x2 subsamples, bilinear.
// Output layout rf[roi][c*49 + bin] f16 (matches head-weight k ordering).
// grid: (192), block: (128) = channels
// ---------------------------------------------------------------------------
__global__ __launch_bounds__(128)
void roi_align_kernel(const _Float16* __restrict__ rel,
                      const float* __restrict__ det1,
                      _Float16* __restrict__ rf) {
    const int roi = blockIdx.x;
    const int c = threadIdx.x;
    const int n = roi / MB;
    const float* d = det1 + (size_t)roi * 10;
    float cx = d[2], cy = d[3], w = d[4], h = d[5];
    float x1 = (cx - 0.5f * w) * 0.125f;
    float y1 = (cy - 0.5f * h) * 0.125f;
    float bw = (w * 0.125f) / 7.f;
    float bh = (h * 0.125f) / 7.f;
    const _Float16* f = rel + ((size_t)n * 128 + c) * HW;

    for (int bin = 0; bin < 49; ++bin) {
        int by = bin / 7, bx = bin - by * 7;
        float s = 0.f;
#pragma unroll
        for (int sy = 0; sy < 2; ++sy) {
#pragma unroll
            for (int sx = 0; sx < 2; ++sx) {
                float yy = y1 + ((float)(by * 2 + sy) + 0.5f) * 0.5f * bh;
                float xx = x1 + ((float)(bx * 2 + sx) + 0.5f) * 0.5f * bw;
                yy = fminf(fmaxf(yy, 0.f), 95.f);
                xx = fminf(fmaxf(xx, 0.f), 95.f);
                int yl = (int)floorf(yy), xl = (int)floorf(xx);
                int yh = min(yl + 1, 95), xh = min(xl + 1, 95);
                float wy = yy - (float)yl, wx = xx - (float)xl;
                float f00 = (float)f[yl * IMW + xl];
                float f01 = (float)f[yl * IMW + xh];
                float f10 = (float)f[yh * IMW + xl];
                float f11 = (float)f[yh * IMW + xh];
                s += f00 * (1.f - wy) * (1.f - wx) + f01 * (1.f - wy) * wx +
                     f10 * wy * (1.f - wx) + f11 * wy * wx;
            }
        }
        rf[(size_t)roi * RFK + c * 49 + bin] = (_Float16)(s * 0.25f);
    }
}

// ---------------------------------------------------------------------------
// Sum pos-channel head weights over spatial: wpsum[q][o] = sum_s W[o][128+q][s]
// ---------------------------------------------------------------------------
__global__ void head_wpsum_kernel(const float* __restrict__ head_w,
                                  float* __restrict__ wpsum) {
    int idx = blockIdx.x * blockDim.x + threadIdx.x;
    if (idx >= 1024) return;
    int q = idx >> 8, o = idx & 255;
    const float* s = head_w + (size_t)o * HLD + (128 + q) * 49;
    float acc = 0.f;
    for (int t = 0; t < 49; ++t) acc += s[t];
    wpsum[q * 256 + o] = acc;
}

// ---------------------------------------------------------------------------
// Pair assembly: x0[p][o] = h1[roi_i][o] + head_b[o] + pos4 . wpsum[:,o]
// Also writes labels/valid. grid: (9216), block: (256)=out features
// ---------------------------------------------------------------------------
__global__ __launch_bounds__(256)
void pair_assemble_kernel(const float* __restrict__ det1,
                          const float* __restrict__ det2,
                          const _Float16* __restrict__ h1,
                          const float* __restrict__ wpsum,
                          const float* __restrict__ head_b,
                          _Float16* __restrict__ x0,
                          float* __restrict__ out_labels,
                          float* __restrict__ out_valid) {
    const int p = blockIdx.x;
    const int n = p / (MB * MB);
    const int r = p - n * MB * MB;
    const int i = r / MB;
    const int j = r - i * MB;
    const float* a = det1 + ((size_t)n * MB + i) * 10;
    const float* b = det2 + ((size_t)n * MB + j) * 10;
    const float eps = 1e-6f;
    float w1 = fmaxf(a[4], eps), h1b = fmaxf(a[5], eps);
    float w2 = fmaxf(b[4], eps), h2b = fmaxf(b[5], eps);
    float pos0 = (b[2] - a[2]) / w1;
    float pos1 = (b[3] - a[3]) / h1b;
    float pos2 = logf(w2 / w1);
    float pos3 = logf(h2b / h1b);

    if (threadIdx.x == 0) {
        out_labels[p] = (a[1] == b[1]) ? 1.f : 0.f;
        out_valid[p]  = (a[0] != -1.f && b[0] != -1.f) ? 1.f : 0.f;
    }
    const int o = threadIdx.x;
    const int roi = n * MB + i;
    float v = (float)h1[(size_t)roi * 256 + o] + head_b[o]
            + pos0 * wpsum[0 * 256 + o] + pos1 * wpsum[1 * 256 + o]
            + pos2 * wpsum[2 * 256 + o] + pos3 * wpsum[3 * 256 + o];
    x0[(size_t)p * 256 + o] = (_Float16)v;
}

// ---------------------------------------------------------------------------
// Classifier: scores[p] = x7[p,:] . cls_w + cls_b  (O=1, plain dot)
// ---------------------------------------------------------------------------
__global__ void cls_kernel(const _Float16* __restrict__ x7,
                           const float* __restrict__ cls_w,
                           const float* __restrict__ cls_b,
                           float* __restrict__ scores) {
    int p = blockIdx.x * blockDim.x + threadIdx.x;
    if (p >= NMM) return;
    float s = cls_b[0];
    for (int k = 0; k < 256; ++k)
        s += (float)x7[(size_t)p * 256 + k] * cls_w[k];
    scores[p] = s;
}

// ---------------------------------------------------------------------------
// Host orchestration
// ---------------------------------------------------------------------------
extern "C" void kernel_launch(void* const* d_in, const int* in_sizes, int n_in,
                              void* d_out, int out_size, void* d_ws, size_t ws_size,
                              hipStream_t stream) {
    const float* corr    = (const float*)d_in[0];
    const float* det1    = (const float*)d_in[2];
    const float* det2    = (const float*)d_in[3];
    const float* w1      = (const float*)d_in[4];
    const float* b1      = (const float*)d_in[5];
    const float* w2      = (const float*)d_in[6];
    const float* b2      = (const float*)d_in[7];
    const float* w3      = (const float*)d_in[8];
    const float* b3      = (const float*)d_in[9];
    const float* head_w  = (const float*)d_in[10];
    const float* head_b  = (const float*)d_in[11];
    const float* fc6_w   = (const float*)d_in[12];
    const float* fc6_b   = (const float*)d_in[13];
    const float* fc7_w   = (const float*)d_in[14];
    const float* fc7_b   = (const float*)d_in[15];
    const float* cls_w   = (const float*)d_in[16];
    const float* cls_b   = (const float*)d_in[17];
    float* out = (float*)d_out;   // [0:9216] scores, [9216:] labels, [18432:] valid

    // ---- workspace layout (bytes, 256-aligned, with aliasing reuse) ----
    char* ws = (char*)d_ws;
    size_t off = 0;
    auto take = [&](size_t bytes) -> char* {
        char* p = ws + off;
        off += (bytes + 255) & ~(size_t)255;
        return p;
    };
    char* corr_region = take((size_t)NB * CORPP * HW * 2);  // corr_h, later c2_h
    char* c1_region   = take((size_t)NB * 256 * HW * 2);    // c1_h, later x0/x6
    char* rel_region  = take((size_t)NB * 128 * HW * 2);    // rel_h, later x7
    _Float16* w1h  = (_Float16*)take((size_t)256 * CORPP * 2);
    _Float16* w2h  = (_Float16*)take((size_t)256 * 2304 * 2);
    _Float16* w3h  = (_Float16*)take((size_t)128 * 2304 * 2);
    _Float16* whh  = (_Float16*)take((size_t)256 * HLD * 2);
    _Float16* w6h  = (_Float16*)take((size_t)256 * 256 * 2);
    _Float16* w7h  = (_Float16*)take((size_t)256 * 256 * 2);
    _Float16* rf_h = (_Float16*)take((size_t)NMP * RFK * 2);   // 256 rows (padded)
    _Float16* h1_h = (_Float16*)take((size_t)NMP * 256 * 2);   // 256 rows (padded)
    float*    wps  = (float*)take(4 * 256 * 4);

    _Float16* corr_h = (_Float16*)corr_region;
    _Float16* c1_h   = (_Float16*)c1_region;
    _Float16* c2_h   = (_Float16*)corr_region;                      // alias: corr dead
    _Float16* rel_h  = (_Float16*)rel_region;
    _Float16* x0_h   = (_Float16*)c1_region;                        // alias: c1 dead
    _Float16* x6_h   = (_Float16*)(c1_region + (size_t)NMM * 256 * 2);
    _Float16* x7_h   = (_Float16*)rel_region;                       // alias: rel dead

    auto cvt = [&](const float* src, _Float16* dst, int K, int Kpad, long long rows) {
        long long total = rows * Kpad;
        int blocks = (int)((total + 255) / 256);
        cvt_pad_kernel<<<blocks, 256, 0, stream>>>(src, dst, K, Kpad, total);
    };

    // ---- precision staging: f32 -> f16 (+ zero-pad K to %32) ----
    cvt(corr,   corr_h, CORP * HW, CORPP * HW, NB);     // pad channels 324->352
    cvt(w1,     w1h, CORP, CORPP, 256);                 // pad K 324->352
    cvt(w2,     w2h, 2304, 2304, 256);
    cvt(w3,     w3h, 2304, 2304, 128);
    cvt(head_w, whh, HLD, HLD, 256);
    cvt(fc6_w,  w6h, 256, 256, 256);
    cvt(fc7_w,  w7h, 256, 256, 256);

    // ---- convc1: 1x1, 352(padded) -> 256, relu ----
    conv_gemm_kernel<1, 1, 0><<<dim3(HW / 64, 256 / 128, NB), 256, 0, stream>>>(
        corr_h, w1h, b1, c1_h, CORPP, 256, 1);
    // ---- convc2: 3x3 pad1, 256 -> 256, relu ----
    conv_gemm_kernel<3, 3, 1><<<dim3(HW / 64, 256 / 128, NB), 256, 0, stream>>>(
        c1_h, w2h, b2, c2_h, 256, 256, 1);
    // ---- conv: 3x3 pad1, 256 -> 128, relu -> rel ----
    conv_gemm_kernel<3, 3, 1><<<dim3(HW / 64, 128 / 128, NB), 256, 0, stream>>>(
        c2_h, w3h, b3, rel_h, 256, 128, 1);

    // ---- ROI align: rel -> rf[192][6272] ----
    roi_align_kernel<<<NM, 128, 0, stream>>>(rel_h, det1, rf_h);

    // ---- head: h1[192][256] = rf * Wr^T (Wr = head_w[:, :128, :, :]);
    //      rows padded to 256 (rows 192..255 are don't-care) ----
    fc_gemm_kernel<<<dim3(NMP / 128, 256 / 64), 256, 0, stream>>>(
        rf_h, RFK, NMP, whh, HLD, nullptr, h1_h, 256, RFK, 0);

    // ---- pos-channel weight sums ----
    head_wpsum_kernel<<<4, 256, 0, stream>>>(head_w, wps);

    // ---- pair assembly: x0[9216][256], labels, valid ----
    pair_assemble_kernel<<<NMM, 256, 0, stream>>>(
        det1, det2, h1_h, wps, head_b, x0_h, out + NMM, out + 2 * NMM);

    // ---- fc6 / fc7 (relu) ----
    fc_gemm_kernel<<<dim3(NMM / 128, 256 / 64), 256, 0, stream>>>(
        x0_h, 256, NMM, w6h, 256, fc6_b, x6_h, 256, 256, 1);
    fc_gemm_kernel<<<dim3(NMM / 128, 256 / 64), 256, 0, stream>>>(
        x6_h, 256, NMM, w7h, 256, fc7_b, x7_h, 256, 256, 1);

    // ---- classifier ----
    cls_kernel<<<(NMM + 255) / 256, 256, 0, stream>>>(x7_h, cls_w, cls_b, out);
}
